// TiMultiHeadAttention_7164005450385
// MI455X (gfx1250) — matrix operand order repaired and checked by reading, not compiled, not run on hardware
//
#include <hip/hip_runtime.h>
#include <hip/hip_bf16.h>

// ---------------------------------------------------------------------------
// TiMultiHeadAttention fused pipeline for MI455X (gfx1250, wave32)
//   B=8, L=200, HID=128, H=4, D=32
// Roofline: tk/tv streams = 328 MB -> ~14us at 23.3 TB/s; all GEMM FLOPs are
// trivial (~0.6 GFLOP).  GEMMs use V_WMMA_F32_16X16X4_F32 (fp32-exact); the
// attention streamer uses b128 loads + width-8 butterflies to stay under the
// per-WGP issue budget and remain purely HBM-bound.
// ---------------------------------------------------------------------------

#define BB   8
#define LL   200
#define HID_ 128
#define NH   4
#define DD   32
#define MM   (BB * LL)          // 1600 rows, 100 tiles of 16
#define EPS_ 1e-5f

typedef float     v2f  __attribute__((ext_vector_type(2)));
typedef float     v8f  __attribute__((ext_vector_type(8)));
typedef _Float16  v16h __attribute__((ext_vector_type(16)));

#if defined(__has_builtin)
#  if __has_builtin(__builtin_amdgcn_wmma_f32_16x16x4_f32)
#    define HAVE_F32_WMMA 1
#  endif
#endif

// Compute one 16x16 tile of A(16xK) * B(KxN) with K=128, fp32 row-major
// inputs.  Per-wave; EXEC must be all ones (callers have no divergence).
__device__ __forceinline__ v8f wmma_gemm_k128(const float* __restrict__ A, int lda,
                                              const float* __restrict__ Bm, int ldb) {
  v8f c = {};
  const int lane = threadIdx.x & 31;
  const int half = lane >> 4;      // 0: lanes 0-15, 1: lanes 16-31
  const int l16  = lane & 15;
#if defined(HAVE_F32_WMMA)
  // ISA 7.12.2: 32-bit A 16x4 -> VGPR0=K(0|2), VGPR1=K(1|3) per lane-half.
  #pragma unroll 4
  for (int k = 0; k < 128; k += 4) {
    v2f a, b;
    const float* ap = A + l16 * lda + k + 2 * half;
    a.x = ap[0];
    a.y = ap[1];
    const float* bp = Bm + (k + 2 * half) * ldb + l16;   // B is KxN row-major
    b.x = bp[0];
    b.y = bp[ldb];
    c = __builtin_amdgcn_wmma_f32_16x16x4_f32(false, a, false, b,
                                              (short)0, c, false, false);
  }
#else
  // Fallback: f16 inputs, f32 accumulate (codegen-confirmed builtin).
  for (int k = 0; k < 128; k += 32) {
    v16h a, b;
    #pragma unroll
    for (int e = 0; e < 16; ++e) {
      const int ka = k + e + (half << 4);
      a[e] = (_Float16)A[l16 * lda + ka];
      const int kb = k + e + (half << 4);
      b[e] = (_Float16)Bm[kb * ldb + l16];
    }
    c = __builtin_amdgcn_wmma_f32_16x16x32_f16(false, a, false, b,
                                               (short)0, c, false, false);
  }
#endif
  return c;
}

// ---------------------------------------------------------------------------
// Kernel 1: QKV projections.  grid=(MM/16, 3), block=256 (8 waves).
// Wave w computes output columns [16w, 16w+16) of matrix blockIdx.y.
// ---------------------------------------------------------------------------
__global__ void ti_qkv_kernel(const float* __restrict__ hidden,
                              const float* __restrict__ Wq, const float* __restrict__ bq,
                              const float* __restrict__ Wk, const float* __restrict__ bk,
                              const float* __restrict__ Wv, const float* __restrict__ bv,
                              float* __restrict__ Q, float* __restrict__ K,
                              float* __restrict__ V) {
  const int mt    = blockIdx.x;            // 16-row tile index
  const int which = blockIdx.y;            // 0=Q 1=K 2=V
  const float* W    = (which == 0) ? Wq : (which == 1) ? Wk : Wv;
  const float* bias = (which == 0) ? bq : (which == 1) ? bk : bv;
  float*       out  = (which == 0) ? Q  : (which == 1) ? K  : V;

  const int wave = threadIdx.x >> 5;
  const int n0   = wave * 16;
  const int lane = threadIdx.x & 31;
  const int half = lane >> 4;
  const int l16  = lane & 15;

  v8f c = wmma_gemm_k128(hidden + (size_t)mt * 16 * HID_, HID_, W + n0, HID_);

  const float bval = bias[n0 + l16];
  #pragma unroll
  for (int r = 0; r < 8; ++r) {
    const int m = mt * 16 + r + 8 * half;      // C layout: VGPR r -> row r+8*half
    out[(size_t)m * HID_ + n0 + l16] = c[r] + bval;
  }
}

// ---------------------------------------------------------------------------
// Kernel 2: streaming attention.  grid=B*L (one block per query row),
// block=128 (4 waves).  Each wave owns one whole j-row per step:
//   lane l -> channels 4l..4l+3, head = l>>3.
// b128 loads, width-8 butterflies; probs normalized in LDS before pass 2.
// ---------------------------------------------------------------------------
__global__ void ti_attn_kernel(const float* __restrict__ Q,
                               const float* __restrict__ K,
                               const float* __restrict__ V,
                               const float* __restrict__ tk,
                               const float* __restrict__ tv,
                               float* __restrict__ ctx) {
  const int bi   = blockIdx.x;          // b*L + i
  const int b    = bi / LL;
  const int i    = bi % LL;
  const int t    = threadIdx.x;         // 0..127
  const int wave = t >> 5;              // 0..3: j-row owner
  const int l    = t & 31;              // lane
  const int hh   = l >> 3;              // head owned by this 8-lane group
  const int c0   = l << 2;              // first of 4 channels

  __shared__ float sc[NH][LL];          // scores -> normalized probs
  __shared__ float accs[4][HID_];       // cross-wave ctx partials

  const float scale  = 0.17677669529663687f;   // 1/sqrt(32)
  const float* tkrow = tk + (size_t)bi * LL * HID_;
  const float* tvrow = tv + (size_t)bi * LL * HID_;
  const float4 q4 = *(const float4*)(Q + (size_t)bi * HID_ + c0);

  // ---- pass 1: scores[j] = (Q . (K[j] + tk[i,j])) * scale, j <= i ----
  for (int j0 = 0; j0 <= i; j0 += 4) {
    const int jj = j0 + wave;           // wave-uniform
    if (jj + 16 <= i)
      __builtin_prefetch(tkrow + (size_t)(jj + 16) * HID_ + c0, 0, 0);
    if (jj <= i) {
      const float4 kv  = *(const float4*)(K + (size_t)(b * LL + jj) * HID_ + c0);
      const float4 tk4 = *(const float4*)(tkrow + (size_t)jj * HID_ + c0);
      float p = q4.x * (kv.x + tk4.x) + q4.y * (kv.y + tk4.y) +
                q4.z * (kv.z + tk4.z) + q4.w * (kv.w + tk4.w);
      #pragma unroll
      for (int off = 4; off; off >>= 1) p += __shfl_xor(p, off, 8);
      if ((l & 7) == 0) sc[hh][jj] = p * scale;
    }
  }
  __syncthreads();

  // ---- softmax over j in [0, i]: wave h handles head h (4 waves, 4 heads) --
  {
    const int h = wave;
    const int d = l;
    float m = -1e30f;
    for (int j = d; j <= i; j += 32) m = fmaxf(m, sc[h][j]);
    #pragma unroll
    for (int off = 16; off; off >>= 1) m = fmaxf(m, __shfl_xor(m, off, 32));
    float s = 0.0f;
    for (int j = d; j <= i; j += 32) {
      const float e = __expf(sc[h][j] - m);
      sc[h][j] = e;
      s += e;
    }
    #pragma unroll
    for (int off = 16; off; off >>= 1) s += __shfl_xor(s, off, 32);
    const float inv = 1.0f / s;
    for (int j = d; j <= i; j += 32) sc[h][j] *= inv;   // normalized probs
  }
  __syncthreads();

  // ---- pass 2: ctx = sum_j p_j * (V[j] + tv[i,j]) ----
  float4 acc = make_float4(0.f, 0.f, 0.f, 0.f);
  for (int j0 = 0; j0 <= i; j0 += 4) {
    const int jj = j0 + wave;           // wave-uniform
    if (jj + 16 <= i)
      __builtin_prefetch(tvrow + (size_t)(jj + 16) * HID_ + c0, 0, 0);
    if (jj <= i) {
      const float  p   = sc[hh][jj];    // broadcast within 8-lane group
      const float4 vv  = *(const float4*)(V + (size_t)(b * LL + jj) * HID_ + c0);
      const float4 tv4 = *(const float4*)(tvrow + (size_t)jj * HID_ + c0);
      acc.x += p * (vv.x + tv4.x);
      acc.y += p * (vv.y + tv4.y);
      acc.z += p * (vv.z + tv4.z);
      acc.w += p * (vv.w + tv4.w);
    }
  }
  *(float4*)&accs[wave][c0] = acc;
  __syncthreads();

  ctx[(size_t)bi * HID_ + t] =
      accs[0][t] + accs[1][t] + accs[2][t] + accs[3][t];
}

// ---------------------------------------------------------------------------
// Kernel 3: out = LayerNorm(ctx @ Wd + bd + hidden).  grid=MM/16, block=256.
// 8 waves compute the 16x128 GEMM tile via WMMA; then 16-lane groups do the
// row-wise mean/var reduction.
// ---------------------------------------------------------------------------
__global__ void ti_out_kernel(const float* __restrict__ ctx,
                              const float* __restrict__ Wd,
                              const float* __restrict__ bd,
                              const float* __restrict__ hidden,
                              const float* __restrict__ gamma,
                              const float* __restrict__ beta,
                              float* __restrict__ out) {
  __shared__ float tile[16][HID_];

  const int mt   = blockIdx.x;
  const int wave = threadIdx.x >> 5;
  const int n0   = wave * 16;
  const int lane = threadIdx.x & 31;
  const int half = lane >> 4;
  const int l16  = lane & 15;

  v8f c = wmma_gemm_k128(ctx + (size_t)mt * 16 * HID_, HID_, Wd + n0, HID_);

  const float bval = bd[n0 + l16];
  #pragma unroll
  for (int r = 0; r < 8; ++r) {
    const int m = r + 8 * half;
    tile[m][n0 + l16] =
        c[r] + bval + hidden[(size_t)(mt * 16 + m) * HID_ + n0 + l16];
  }
  __syncthreads();

  // LayerNorm: 16 rows x 16 lanes, 8 channels per lane (stride 16).
  const int row = threadIdx.x >> 4;
  const int cp  = threadIdx.x & 15;
  float sum = 0.0f, sq = 0.0f;
  #pragma unroll
  for (int e = 0; e < 8; ++e) {
    const float v = tile[row][cp + e * 16];
    sum += v;
    sq  += v * v;
  }
  #pragma unroll
  for (int off = 8; off; off >>= 1) {
    sum += __shfl_xor(sum, off, 16);
    sq  += __shfl_xor(sq,  off, 16);
  }
  const float mu  = sum * (1.0f / HID_);
  const float var = sq * (1.0f / HID_) - mu * mu;
  const float rs  = rsqrtf(var + EPS_);
  #pragma unroll
  for (int e = 0; e < 8; ++e) {
    const int cidx = cp + e * 16;
    const float v  = (tile[row][cidx] - mu) * rs;
    out[(size_t)(mt * 16 + row) * HID_ + cidx] = v * gamma[cidx] + beta[cidx];
  }
}

// ---------------------------------------------------------------------------
extern "C" void kernel_launch(void* const* d_in, const int* in_sizes, int n_in,
                              void* d_out, int out_size, void* d_ws, size_t ws_size,
                              hipStream_t stream) {
  (void)in_sizes; (void)n_in; (void)out_size; (void)ws_size;

  const float* hidden = (const float*)d_in[0];
  // d_in[1] = attention_mask (bool, causal) -- implied by j <= i loop
  const float* tk     = (const float*)d_in[2];
  const float* tv     = (const float*)d_in[3];
  const float* Wq     = (const float*)d_in[4];
  const float* bq     = (const float*)d_in[5];
  const float* Wk     = (const float*)d_in[6];
  const float* bk     = (const float*)d_in[7];
  const float* Wv     = (const float*)d_in[8];
  const float* bv     = (const float*)d_in[9];
  const float* Wd     = (const float*)d_in[10];
  const float* bd     = (const float*)d_in[11];
  const float* gamma  = (const float*)d_in[12];
  const float* beta   = (const float*)d_in[13];
  float* out = (float*)d_out;

  const size_t mat = (size_t)MM * HID_;   // 204800 floats per buffer
  float* Qb  = (float*)d_ws;
  float* Kb  = Qb + mat;
  float* Vb  = Kb + mat;
  float* Cb  = Vb + mat;                  // total 3.3 MB of workspace

  ti_qkv_kernel<<<dim3(MM / 16, 3), 256, 0, stream>>>(hidden, Wq, bq, Wk, bk,
                                                      Wv, bv, Qb, Kb, Vb);
  ti_attn_kernel<<<dim3(MM), 128, 0, stream>>>(Qb, Kb, Vb, tk, tv, Cb);
  ti_out_kernel<<<dim3(MM / 16), 256, 0, stream>>>(Cb, Wd, bd, hidden,
                                                   gamma, beta, out);
}
